// MultiHeadSelfAttention_63642825392602
// MI455X (gfx1250) — matrix operand order repaired and checked by reading, not compiled, not run on hardware
//
#include <hip/hip_runtime.h>
#include <hip/hip_bf16.h>

#define B_ 8
#define S_ 1024
#define E_ 768
#define H_ 12
#define D_ 64
#define M_ (B_*S_)   // 8192 rows in the flattened [B*S, E] activations

typedef __bf16 bf16;
typedef __attribute__((ext_vector_type(8)))  __bf16 v8bf;
typedef __attribute__((ext_vector_type(16))) __bf16 v16bf;
typedef __attribute__((ext_vector_type(8)))  float  v8f;

static __device__ __forceinline__ v8f wmma_bf16(v16bf a, v16bf b, v8f c) {
  // D = A(16x32 bf16) * B(32x16 bf16) + C(16x16 f32)
  return __builtin_amdgcn_wmma_f32_16x16x32_bf16(false, a, false, b, (short)0, c, false, false);
}

static __device__ __forceinline__ v8f v8f_zero() {
  v8f z;
#pragma unroll
  for (int i = 0; i < 8; ++i) z[i] = 0.f;
  return z;
}

static __device__ __forceinline__ v16bf cat8(v8bf lo, v8bf hi) {
  return __builtin_shufflevector(lo, hi, 0,1,2,3,4,5,6,7,8,9,10,11,12,13,14,15);
}

// CDNA5 async memory->LDS copy (ASYNCcnt-tracked).  GV mode: per-lane LDS byte
// address in VDST, 64-bit global address in a VGPR pair, SADDR=off.
static __device__ __forceinline__ void async_ld_b128(const void* lds_ptr,
                                                     const void* gptr) {
  uint32_t loff = (uint32_t)(uintptr_t)lds_ptr;   // generic LDS addr bits[31:0]
  uint64_t g = (uint64_t)(uintptr_t)gptr;
  asm volatile("global_load_async_to_lds_b128 %0, %1, off"
               :: "v"(loff), "v"(g) : "memory");
}
static __device__ __forceinline__ void wait_async0() {
  asm volatile("s_wait_asynccnt 0" ::: "memory");
}

// A-fragment (16x32, 16-bit): lanes 0-15 -> row M=lane, K chunks {0-7,16-23};
// lanes 16-31 -> row M=lane-16, K chunks {8-15,24-31}.  Two 16B loads.
static __device__ __forceinline__ v16bf load_afrag(const bf16* base, int ld,
                                                   int rowBase, int kBase, int lane) {
  const bf16* p = base + (rowBase + (lane & 15)) * ld + kBase + ((lane < 16) ? 0 : 8);
  return cat8(*(const v8bf*)p, *(const v8bf*)(p + 16));
}

// B-fragment (32x16, 16-bit): lanes 0-15 -> col N=lane, K chunks {0-7,16-23}; lanes 16-31 ->
// col N=lane-16, K chunks {8-15,24-31}.  Storage is column-contiguous over K ("Bt" rows).
static __device__ __forceinline__ v16bf load_bfrag(const bf16* base, int ld,
                                                   int colBase, int kBase, int lane) {
  const bf16* p = base + (colBase + (lane & 15)) * ld + kBase + ((lane < 16) ? 0 : 8);
  return cat8(*(const v8bf*)p, *(const v8bf*)(p + 16));
}

// ---------------------------------------------------------------------------
// prep: fp32 -> bf16 activation copy; weight transpose to k-contiguous bf16
// ---------------------------------------------------------------------------
__global__ __launch_bounds__(256) void prep_kernel(
    const float* __restrict__ X,
    const float* __restrict__ Wq, const float* __restrict__ Wk,
    const float* __restrict__ Wv, const float* __restrict__ Wo,
    bf16* __restrict__ Xb,
    bf16* __restrict__ Wqt, bf16* __restrict__ Wkt,
    bf16* __restrict__ Wvt, bf16* __restrict__ Wot) {
  const int stride = gridDim.x * blockDim.x;
  const int i0 = blockIdx.x * blockDim.x + threadIdx.x;
  for (int t = i0; t < M_ * E_; t += stride) Xb[t] = (bf16)X[t];
  for (int t = i0; t < E_ * E_; t += stride) {
    int n = t / E_, k = t - n * E_;
    int src = k * E_ + n;          // Wt[n][k] = W[k][n]
    Wqt[t] = (bf16)Wq[src];
    Wkt[t] = (bf16)Wk[src];
    Wvt[t] = (bf16)Wv[src];
    Wot[t] = (bf16)Wo[src];
  }
}

// ---------------------------------------------------------------------------
// Shared 128x128 GEMM mainloop: C[128x128] += A[128xK] * Bt[128xK]^T
// 8 waves in a 4x2 grid, each wave 32x64 = 2x4 WMMA fragments, K-step 32,
// double-buffered LDS filled by async-to-LDS copies (ASYNCcnt), one barrier
// per K-step.
// ---------------------------------------------------------------------------
struct GemmLds {
  bf16 A[2][128][40];   // 32 K + 8 pad (80B rows, 16B-aligned chunks)
  bf16 Bm[2][128][40];
};                      // 40 KB

static __device__ __forceinline__ void gemm_issue_tile(
    const bf16* __restrict__ Ag, const bf16* __restrict__ Bg,
    int rowBase, int colBase, int kBase, GemmLds& lds, int buf, int tid) {
#pragma unroll
  for (int c = 0; c < 2; ++c) {
    int ch = tid * 2 + c;   // 512 chunks of 8 bf16 cover 128x32
    int row = ch >> 2, kc = (ch & 3) * 8;
    async_ld_b128(&lds.A[buf][row][kc],  Ag + (rowBase + row) * E_ + kBase + kc);
    async_ld_b128(&lds.Bm[buf][row][kc], Bg + (colBase + row) * E_ + kBase + kc);
  }
}

static __device__ __forceinline__ void gemm_mainloop(
    const bf16* __restrict__ Ag, const bf16* __restrict__ Bg,
    int rowBase, int colBase, GemmLds& lds, v8f acc[2][4]) {
  const int tid = threadIdx.x, lane = tid & 31, wave = tid >> 5;
  const int waveM = (wave >> 1) * 32, waveN = (wave & 1) * 64;
#pragma unroll
  for (int i = 0; i < 2; ++i)
#pragma unroll
    for (int j = 0; j < 4; ++j) acc[i][j] = v8f_zero();

  gemm_issue_tile(Ag, Bg, rowBase, colBase, 0, lds, 0, tid);
  wait_async0();
  __syncthreads();
  int buf = 0;
#pragma unroll 1
  for (int kt = 0; kt < E_ / 32; ++kt) {
    if (kt + 1 < E_ / 32)
      gemm_issue_tile(Ag, Bg, rowBase, colBase, (kt + 1) * 32, lds, buf ^ 1, tid);
    v16bf af[2], bfr[4];
    af[0] = load_afrag(&lds.A[buf][0][0], 40, waveM,      0, lane);
    af[1] = load_afrag(&lds.A[buf][0][0], 40, waveM + 16, 0, lane);
#pragma unroll
    for (int j = 0; j < 4; ++j)
      bfr[j] = load_bfrag(&lds.Bm[buf][0][0], 40, waveN + j * 16, 0, lane);
#pragma unroll
    for (int i = 0; i < 2; ++i)
#pragma unroll
      for (int j = 0; j < 4; ++j) acc[i][j] = wmma_bf16(af[i], bfr[j], acc[i][j]);
    if (kt + 1 < E_ / 32) {
      wait_async0();      // this wave's async fills of buf^1 are done
      __syncthreads();    // everyone's fills done, everyone's reads of buf done
      buf ^= 1;
    }
  }
}

// ---------------------------------------------------------------------------
// QKV projection: grid.z selects q/k/v. Bias folded; 1/sqrt(D) folded into Q.
// Q,K written [B,H,S,D] bf16 (d-contiguous); V written transposed [B,H,D,S].
// ---------------------------------------------------------------------------
__global__ __launch_bounds__(256) void qkv_gemm_kernel(
    const bf16* __restrict__ Xb,
    const bf16* __restrict__ Wqt, const bf16* __restrict__ Wkt, const bf16* __restrict__ Wvt,
    const float* __restrict__ bq, const float* __restrict__ bk, const float* __restrict__ bv,
    bf16* __restrict__ Q, bf16* __restrict__ K, bf16* __restrict__ V) {
  __shared__ GemmLds lds;
  const int which = blockIdx.z;
  const bf16* Bg = (which == 0) ? Wqt : (which == 1 ? Wkt : Wvt);
  const float* bias = (which == 0) ? bq : (which == 1 ? bk : bv);
  const int rowBase = blockIdx.y * 128, colBase = blockIdx.x * 128;
  v8f acc[2][4];
  gemm_mainloop(Xb, Bg, rowBase, colBase, lds, acc);

  const int tid = threadIdx.x, lane = tid & 31, wave = tid >> 5;
  const int waveM = (wave >> 1) * 32, waveN = (wave & 1) * 64;
  const float scale = (which == 0) ? 0.125f : 1.0f;   // 1/sqrt(64) on Q
#pragma unroll
  for (int i = 0; i < 2; ++i)
#pragma unroll
    for (int j = 0; j < 4; ++j)
#pragma unroll
      for (int r = 0; r < 8; ++r) {
        int m = rowBase + waveM + i * 16 + ((lane < 16) ? r : r + 8);
        int n = colBase + waveN + j * 16 + (lane & 15);
        float val = (acc[i][j][r] + bias[n]) * scale;
        int b = m >> 10, s = m & (S_ - 1);
        int h = n >> 6,  d = n & (D_ - 1);
        bf16 ov = (bf16)val;
        if (which == 2)
          V[(size_t)(((b * H_ + h) * D_ + d) << 10) + s] = ov;        // [B,H,D,S]
        else {
          bf16* dst = (which == 0) ? Q : K;
          dst[(size_t)(((b * H_ + h) << 10) + s) * D_ + d] = ov;      // [B,H,S,D]
        }
      }
}

// ---------------------------------------------------------------------------
// Flash attention: one workgroup per (128-query tile, head). 8 waves; each
// wave owns 16 complete query rows -> softmax reductions stay within 16 lanes.
// K/V tiles are filled with async-to-LDS copies.
// ---------------------------------------------------------------------------
struct AttnLds {
  bf16 Kt[128][72];     // key block, row = key, contiguous over d   (18.0 KB)
  bf16 Vt[64][136];     // V^T block, row = d, contiguous over keys  (17.0 KB)
  bf16 P[8][16][40];    // per-wave P staging (C-layout -> A-layout) (10.0 KB)
};

__global__ __launch_bounds__(256) void attn_kernel(
    const bf16* __restrict__ Q, const bf16* __restrict__ K,
    const bf16* __restrict__ V, bf16* __restrict__ Ab) {
  __shared__ AttnLds lds;
  const int tid = threadIdx.x, lane = tid & 31, wave = tid >> 5;
  const int bh = blockIdx.y;                 // 0..95
  const int qBase = blockIdx.x * 128;
  const int bb = bh / H_, hh = bh - bb * H_;
  const bf16* Qh = Q + (size_t)bh * S_ * D_;
  const bf16* Kh = K + (size_t)bh * S_ * D_;
  const bf16* Vh = V + (size_t)bh * D_ * S_;

  // Load this wave's 16 Q rows as two A-fragments (k-steps over D=64).
  const int qRow = qBase + wave * 16;
  v16bf aq[2];
#pragma unroll
  for (int kc = 0; kc < 2; ++kc) {
    const bf16* p = Qh + (qRow + (lane & 15)) * D_ + kc * 32 + ((lane < 16) ? 0 : 8);
    aq[kc] = cat8(*(const v8bf*)p, *(const v8bf*)(p + 16));
  }

  v8f o[4];
#pragma unroll
  for (int j = 0; j < 4; ++j) o[j] = v8f_zero();
  v8f m, l;
#pragma unroll
  for (int r = 0; r < 8; ++r) { m[r] = -1e30f; l[r] = 0.f; }

#pragma unroll 1
  for (int kb = 0; kb < S_ / 128; ++kb) {
    const int keyBase = kb * 128;
    __syncthreads();   // previous iteration's Kt/Vt readers are done
#pragma unroll
    for (int c = 0; c < 4; ++c) {          // 128x64 key block -> Kt (async)
      int ch = tid * 4 + c;
      int row = ch >> 3, kc = (ch & 7) * 8;
      async_ld_b128(&lds.Kt[row][kc], Kh + (keyBase + row) * D_ + kc);
    }
#pragma unroll
    for (int c = 0; c < 4; ++c) {          // 64x128 V^T block -> Vt (async)
      int ch = tid * 4 + c;
      int row = ch >> 4, kc = (ch & 15) * 8;
      async_ld_b128(&lds.Vt[row][kc], Vh + row * S_ + keyBase + kc);
    }
    wait_async0();
    __syncthreads();

    // S = Q * K^T  (16 queries x 128 keys per wave; Q pre-scaled by 1/sqrt(D))
    v8f sc[8];
#pragma unroll
    for (int f = 0; f < 8; ++f) sc[f] = v8f_zero();
#pragma unroll
    for (int kc = 0; kc < 2; ++kc)
#pragma unroll
      for (int f = 0; f < 8; ++f)
        sc[f] = wmma_bf16(aq[kc],
                          load_bfrag(&lds.Kt[0][0], 72, f * 16, kc * 32, lane),
                          sc[f]);

    // online softmax: row max over this key block (reduce frags, then lanes)
    v8f mp = sc[0];
#pragma unroll
    for (int f = 1; f < 8; ++f)
#pragma unroll
      for (int r = 0; r < 8; ++r) mp[r] = fmaxf(mp[r], sc[f][r]);
#pragma unroll
    for (int off = 1; off < 16; off <<= 1)
#pragma unroll
      for (int r = 0; r < 8; ++r) mp[r] = fmaxf(mp[r], __shfl_xor(mp[r], off, 32));

    v8f alpha;
#pragma unroll
    for (int r = 0; r < 8; ++r) {
      float mn = fmaxf(m[r], mp[r]);
      alpha[r] = __expf(m[r] - mn);
      m[r] = mn;
    }

    v8f ls = v8f_zero();
#pragma unroll
    for (int f = 0; f < 8; ++f)
#pragma unroll
      for (int r = 0; r < 8; ++r) {
        float p = __expf(sc[f][r] - m[r]);
        sc[f][r] = p;
        ls[r] += p;
      }
#pragma unroll
    for (int off = 1; off < 16; off <<= 1)
#pragma unroll
      for (int r = 0; r < 8; ++r) ls[r] += __shfl_xor(ls[r], off, 32);
#pragma unroll
    for (int r = 0; r < 8; ++r) l[r] = l[r] * alpha[r] + ls[r];
#pragma unroll
    for (int j = 0; j < 4; ++j)
#pragma unroll
      for (int r = 0; r < 8; ++r) o[j][r] *= alpha[r];

    // O += P * V : stage 16x32 P chunks through per-wave LDS (C->A transpose)
#pragma unroll
    for (int kc = 0; kc < 4; ++kc) {
#pragma unroll
      for (int ff = 0; ff < 2; ++ff) {
        int f = kc * 2 + ff;
#pragma unroll
        for (int r = 0; r < 8; ++r) {
          int row = (lane < 16) ? r : (r + 8);
          lds.P[wave][row][ff * 16 + (lane & 15)] = (bf16)sc[f][r];
        }
      }
      asm volatile("s_wait_dscnt 0" ::: "memory");   // per-wave LDS RAW fence
      v16bf ap = load_afrag(&lds.P[wave][0][0], 40, 0, 0, lane);
#pragma unroll
      for (int j = 0; j < 4; ++j)
        o[j] = wmma_bf16(ap,
                         load_bfrag(&lds.Vt[0][0], 136, j * 16, kc * 32, lane),
                         o[j]);
    }
  }

  // finalize (O / l) and write concat layout [B,S,E] in bf16
#pragma unroll
  for (int j = 0; j < 4; ++j)
#pragma unroll
    for (int r = 0; r < 8; ++r) {
      int s = qRow + ((lane < 16) ? r : r + 8);
      int e = hh * D_ + j * 16 + (lane & 15);
      Ab[((size_t)(bb * S_ + s)) * E_ + e] = (bf16)(o[j][r] / l[r]);
    }
}

// ---------------------------------------------------------------------------
// Output projection: out = concat @ Wo + bo, fp32 result straight to d_out.
// ---------------------------------------------------------------------------
__global__ __launch_bounds__(256) void oproj_kernel(
    const bf16* __restrict__ Ab, const bf16* __restrict__ Wot,
    const float* __restrict__ bo, float* __restrict__ out) {
  __shared__ GemmLds lds;
  const int rowBase = blockIdx.y * 128, colBase = blockIdx.x * 128;
  v8f acc[2][4];
  gemm_mainloop(Ab, Wot, rowBase, colBase, lds, acc);

  const int tid = threadIdx.x, lane = tid & 31, wave = tid >> 5;
  const int waveM = (wave >> 1) * 32, waveN = (wave & 1) * 64;
#pragma unroll
  for (int i = 0; i < 2; ++i)
#pragma unroll
    for (int j = 0; j < 4; ++j)
#pragma unroll
      for (int r = 0; r < 8; ++r) {
        int mg = rowBase + waveM + i * 16 + ((lane < 16) ? r : r + 8);
        int n  = colBase + waveN + j * 16 + (lane & 15);
        out[(size_t)mg * E_ + n] = acc[i][j][r] + bo[n];
      }
}

// ---------------------------------------------------------------------------
extern "C" void kernel_launch(void* const* d_in, const int* in_sizes, int n_in,
                              void* d_out, int out_size, void* d_ws, size_t ws_size,
                              hipStream_t stream) {
  const float* X  = (const float*)d_in[0];
  const float* Wq = (const float*)d_in[1];
  const float* bq = (const float*)d_in[2];
  const float* Wk = (const float*)d_in[3];
  const float* bk = (const float*)d_in[4];
  const float* Wv = (const float*)d_in[5];
  const float* bv = (const float*)d_in[6];
  const float* Wo = (const float*)d_in[7];
  const float* bo = (const float*)d_in[8];
  float* out = (float*)d_out;

  bf16* ws  = (bf16*)d_ws;
  bf16* Xb  = ws;                              // [8192 x 768]
  bf16* Wqt = Xb  + (size_t)M_ * E_;           // [768 x 768] (k-contiguous)
  bf16* Wkt = Wqt + (size_t)E_ * E_;
  bf16* Wvt = Wkt + (size_t)E_ * E_;
  bf16* Wot = Wvt + (size_t)E_ * E_;
  bf16* Qh  = Wot + (size_t)E_ * E_;           // [B,H,S,D]
  bf16* Kh  = Qh  + (size_t)M_ * E_;           // [B,H,S,D]
  bf16* Vh  = Kh  + (size_t)M_ * E_;           // [B,H,D,S]
  bf16* Ab  = Vh  + (size_t)M_ * E_;           // [B,S,E] concat
  // total ws use: ~67.6 MB

  hipLaunchKernelGGL(prep_kernel, dim3(1024), dim3(256), 0, stream,
                     X, Wq, Wk, Wv, Wo, Xb, Wqt, Wkt, Wvt, Wot);
  hipLaunchKernelGGL(qkv_gemm_kernel, dim3(E_ / 128, M_ / 128, 3), dim3(256), 0, stream,
                     Xb, Wqt, Wkt, Wvt, bq, bk, bv, Qh, Kh, Vh);
  hipLaunchKernelGGL(attn_kernel, dim3(S_ / 128, B_ * H_), dim3(256), 0, stream,
                     Qh, Kh, Vh, Ab);
  hipLaunchKernelGGL(oproj_kernel, dim3(E_ / 128, M_ / 128), dim3(256), 0, stream,
                     Ab, Wot, bo, out);
}
